// ContrastiveLoss_35304631174136
// MI455X (gfx1250) — compile-verified, matrix operand order checked by reading, not run on hardware
//
#include <hip/hip_runtime.h>
#include <math.h>

typedef _Float16 v16h __attribute__((ext_vector_type(16)));
typedef _Float16 v8h  __attribute__((ext_vector_type(8)));
typedef float    v8f  __attribute__((ext_vector_type(8)));

#define B_ 64
#define C_ 512
#define D_ 256
#define TEMP_INV 2.0f   // 1 / 0.5

union V16 { v16h v; v8h h[2]; };

// ---------------------------------------------------------------------------
// Zero the scalar accumulator (ws is poisoned, not zeroed, by the harness).
// ---------------------------------------------------------------------------
__global__ void zero_acc_kernel(float* acc) { acc[0] = 0.0f; }

// ---------------------------------------------------------------------------
// L2-normalize each D=256 row and convert to f16. One wave32 per row.
// ---------------------------------------------------------------------------
__global__ __launch_bounds__(256) void norm_convert_kernel(
    const float* __restrict__ src, _Float16* __restrict__ dst, int nrows) {
  const int wave = threadIdx.x >> 5;
  const int lane = threadIdx.x & 31;
  const int row  = blockIdx.x * 8 + wave;
  if (row >= nrows) return;

  const float* p = src + (size_t)row * D_ + lane * 8;
  float4 x0 = *(const float4*)(p);
  float4 x1 = *(const float4*)(p + 4);
  float ss = x0.x*x0.x + x0.y*x0.y + x0.z*x0.z + x0.w*x0.w
           + x1.x*x1.x + x1.y*x1.y + x1.z*x1.z + x1.w*x1.w;
#pragma unroll
  for (int m = 16; m >= 1; m >>= 1) ss += __shfl_xor(ss, m, 32);
  const float inv = 1.0f / fmaxf(sqrtf(ss), 1e-12f);

  v8h o;
  o[0] = (_Float16)(x0.x * inv); o[1] = (_Float16)(x0.y * inv);
  o[2] = (_Float16)(x0.z * inv); o[3] = (_Float16)(x0.w * inv);
  o[4] = (_Float16)(x1.x * inv); o[5] = (_Float16)(x1.y * inv);
  o[6] = (_Float16)(x1.z * inv); o[7] = (_Float16)(x1.w * inv);
  *(v8h*)(dst + (size_t)row * D_ + lane * 8) = o;
}

// ---------------------------------------------------------------------------
// Batched GEMM (WMMA f16 -> f32) fused with the contrastive-loss reduction.
// Block = 4 waves. blockIdx.x = c-tile (16 rows), blockIdx.y = batch.
// Wave w covers k-columns [w*128, w*128+128) as 8 tiles of 16.
// B fragments are double-buffered (prefetch tile kt+1 while computing kt);
// the K=256 reduction runs as two independent 4-deep WMMA chains.
// ---------------------------------------------------------------------------
__global__ __launch_bounds__(128) void gemm_loss_kernel(
    const _Float16* __restrict__ vh, const _Float16* __restrict__ th,
    float* __restrict__ acc_out) {
  const int ct   = blockIdx.x;          // c-tile index (0..31)
  const int b    = blockIdx.y;          // batch       (0..63)
  const int wave = threadIdx.x >> 5;    // 0..3
  const int lane = threadIdx.x & 31;
  const int n    = lane & 15;           // fragment row/col within 16
  const int sel  = lane >> 4;           // which K-half pattern this lane owns

  // ---- A fragments: 16 rows of V-hat, register resident for all k-tiles.
  const _Float16* vbase = vh + ((size_t)b * C_ + (size_t)ct * 16 + n) * D_;
  V16 a[8];
#pragma unroll
  for (int kc = 0; kc < 8; ++kc) {
    const int dbase = kc * 32;
    a[kc].h[0] = *(const v8h*)(vbase + dbase + sel * 8);
    a[kc].h[1] = *(const v8h*)(vbase + dbase + 16 + sel * 8);
  }

  float rs[8], sp[8];
#pragma unroll
  for (int r = 0; r < 8; ++r) { rs[r] = 0.0f; sp[r] = 0.0f; }

  const _Float16* tbatch = th + (size_t)b * C_ * D_;
  const int kt0 = wave * 8;

  // ---- Double-buffered B fragments.
  V16 bf[2][8];
  {
    const _Float16* tb = tbatch + ((size_t)kt0 * 16 + n) * D_;
#pragma unroll
    for (int kc = 0; kc < 8; ++kc) {
      bf[0][kc].h[0] = *(const v8h*)(tb + kc * 32 + sel * 8);
      bf[0][kc].h[1] = *(const v8h*)(tb + kc * 32 + 16 + sel * 8);
    }
  }

#pragma unroll
  for (int i = 0; i < 8; ++i) {
    const int kt  = kt0 + i;
    const int cur = i & 1;
    const int nxt = cur ^ 1;

    // Prefetch tile kt+1 (compile-time guard under full unroll: no divergence).
    if (i < 7) {
      const _Float16* tb = tbatch + ((size_t)(kt + 1) * 16 + n) * D_;
#pragma unroll
      for (int kc = 0; kc < 8; ++kc) {
        bf[nxt][kc].h[0] = *(const v8h*)(tb + kc * 32 + sel * 8);
        bf[nxt][kc].h[1] = *(const v8h*)(tb + kc * 32 + 16 + sel * 8);
      }
    }

    // Two independent accumulation chains over the K=256 reduction.
    v8f c0 = {}, c1 = {};
#pragma unroll
    for (int kc = 0; kc < 8; kc += 2) {
      c0 = __builtin_amdgcn_wmma_f32_16x16x32_f16(
               false, a[kc].v,     false, bf[cur][kc].v,     (short)0, c0,
               false, false);
      c1 = __builtin_amdgcn_wmma_f32_16x16x32_f16(
               false, a[kc + 1].v, false, bf[cur][kc + 1].v, (short)0, c1,
               false, false);
    }

    // C/D layout: VGPR r, lanes 0-15 -> M=r ; lanes 16-31 -> M=r+8 ; N=lane%16.
    const bool dtile = (kt == ct);
#pragma unroll
    for (int r = 0; r < 8; ++r) {
      const float s = c0[r] + c1[r];
      rs[r] += expf(TEMP_INV * s);
      sp[r] += (dtile && (n == r + sel * 8)) ? s : 0.0f;
    }
  }

  // Reduce across the 16 lanes sharing each row (xor masks stay in-half).
#pragma unroll
  for (int m = 1; m <= 8; m <<= 1) {
#pragma unroll
    for (int r = 0; r < 8; ++r) {
      rs[r] += __shfl_xor(rs[r], m, 32);
      sp[r] += __shfl_xor(sp[r], m, 32);
    }
  }

  __shared__ float lds_rs[4][16];
  __shared__ float lds_sp[4][16];
  if (n == 0) {  // lanes 0 (rows 0-7) and 16 (rows 8-15)
#pragma unroll
    for (int r = 0; r < 8; ++r) {
      lds_rs[wave][r + sel * 8] = rs[r];
      lds_sp[wave][r + sel * 8] = sp[r];
    }
  }
  __syncthreads();

  if (threadIdx.x < 16) {
    const int row = threadIdx.x;
    const float R = lds_rs[0][row] + lds_rs[1][row] + lds_rs[2][row] + lds_rs[3][row];
    const float P = lds_sp[0][row] + lds_sp[1][row] + lds_sp[2][row] + lds_sp[3][row];
    // loss = -log(pos / (pos+neg)) = log(sum_k exp(s/T)) - s_pos/T
    float loss = logf(R) - TEMP_INV * P;
#pragma unroll
    for (int m = 1; m <= 8; m <<= 1) loss += __shfl_xor(loss, m, 32);
    if (row == 0) atomicAdd(acc_out, loss);
  }
}

__global__ void finalize_kernel(const float* __restrict__ acc,
                                float* __restrict__ out) {
  out[0] = acc[0] * (1.0f / (float)(B_ * C_));
}

// ---------------------------------------------------------------------------
extern "C" void kernel_launch(void* const* d_in, const int* in_sizes, int n_in,
                              void* d_out, int out_size, void* d_ws, size_t ws_size,
                              hipStream_t stream) {
  const float* vis = (const float*)d_in[0];
  const float* txt = (const float*)d_in[1];
  float* out = (float*)d_out;

  const size_t elems = (size_t)B_ * C_ * D_;
  _Float16* vh  = (_Float16*)d_ws;
  _Float16* th  = vh + elems;
  float*    acc = (float*)(th + elems);

  zero_acc_kernel<<<1, 1, 0, stream>>>(acc);

  const int nrows = B_ * C_;
  norm_convert_kernel<<<(nrows + 7) / 8, 256, 0, stream>>>(vis, vh, nrows);
  norm_convert_kernel<<<(nrows + 7) / 8, 256, 0, stream>>>(txt, th, nrows);

  dim3 grid(C_ / 16, B_);   // 32 c-tiles x 64 batches = 2048 workgroups
  gemm_loss_kernel<<<grid, 128, 0, stream>>>(vh, th, acc);

  finalize_kernel<<<1, 1, 0, stream>>>(acc, out);
}
